// BrenierMapNN_73023033966916
// MI455X (gfx1250) — matrix operand order repaired and checked by reading, not compiled, not run on hardware
//
#include <hip/hip_runtime.h>
#include <math.h>

// ---------------------------------------------------------------------------
// BrenierMapNN fused forward+backward (grad wrt u) for MI455X / gfx1250.
// All matmuls via v_wmma_f32_16x16x32_bf16 (wave32).
// Each wave owns 32 batch rows (2 row-tiles) so every B fragment is reused
// twice and each wave carries two independent WMMA chains.
// ---------------------------------------------------------------------------

#define ALPHA 0.2f

typedef __attribute__((ext_vector_type(16))) __bf16 bf16x16;
typedef __attribute__((ext_vector_type(8)))  float  f32x8;

union BF16Frag { bf16x16 v; uint4 q[2]; };

static constexpr int Bsz = 65536;
static constexpr int Dd  = 64;
static constexpr int Hh  = 512;

// packed weight region layout (units: halfs) inside d_ws
static constexpr int WU_SZ = Hh * Dd;   // 32768
static constexpr int WZ_SZ = Hh * Hh;   // 262144
static constexpr int OFF_WU0F = 0;
static constexpr int OFF_WU1F = OFF_WU0F + WU_SZ;
static constexpr int OFF_WU2F = OFF_WU1F + WU_SZ;
static constexpr int OFF_WU3F = OFF_WU2F + WU_SZ;
static constexpr int OFF_WZ1F = OFF_WU3F + WU_SZ;
static constexpr int OFF_WZ2F = OFF_WZ1F + WZ_SZ;
static constexpr int OFF_WZ3F = OFF_WZ2F + WZ_SZ;
static constexpr int OFF_WZ1T = OFF_WZ3F + WZ_SZ;
static constexpr int OFF_WZ2T = OFF_WZ1T + WZ_SZ;
static constexpr int OFF_WZ3T = OFF_WZ2T + WZ_SZ;
static constexpr int OFF_WU0T = OFF_WZ3T + WZ_SZ;
static constexpr int OFF_WU1T = OFF_WU0T + WU_SZ;
static constexpr int OFF_WU2T = OFF_WU1T + WU_SZ;
static constexpr int OFF_WU3T = OFF_WU2T + WU_SZ;
static constexpr int W_END    = OFF_WU3T + WU_SZ;   // 1,835,008 halfs = 3.5 MB

#define NW 4                 // waves per block
#define MT (NW * 32)         // 128 batch rows per block (32 rows per wave)

// ---------------------------------------------------------------------------
// helpers
// ---------------------------------------------------------------------------
static __device__ __forceinline__ f32x8 wmma_bf16(bf16x16 a, bf16x16 b, f32x8 c) {
  return __builtin_amdgcn_wmma_f32_16x16x32_bf16(false, a, false, b, (short)0, c,
                                                 false, false);
}

// A fragment (16x32 bf16) from LDS, row-major with `stride` halfs per row.
// ISA layout: lanes 0-15 hold K 0-7 (v0-3) and 16-23 (v4-7); lanes 16-31 hold
// K 8-15 and 24-31 -> exactly two 16B loads at col {kc*32+hi*8, +16}.
static __device__ __forceinline__ bf16x16 ldA(const __bf16* base, int stride,
                                              int lane, int kc) {
  const int row = lane & 15, hi = lane >> 4;
  const __bf16* p = base + row * stride + kc * 32 + hi * 8;
  BF16Frag f;
  f.q[0] = *(const uint4*)(p);
  f.q[1] = *(const uint4*)(p + 16);
  return f.v;
}

// B fragment: pre-packed block of 32 lanes x 16 contiguous halfs (1 KB).
static __device__ __forceinline__ bf16x16 ldB(const __bf16* blk, int lane) {
  const uint4* p = (const uint4*)blk + (lane << 1);
  BF16Frag f;
  f.q[0] = p[0];
  f.q[1] = p[1];
  return f.v;
}

// ---------------------------------------------------------------------------
// weight prep: exp() + bf16 + WMMA B-fragment blocking.
// dst block = (ntile, kc); within a block: lane (0..31) * 16 halfs.
// element: n = t*16 + (lane&15) ; k = kc*32 + 8*(lane>>4) + (h&7) + 16*(h>>3)
// value  = exp(src[n*sN + k*sK])   (sN/sK give optional transpose)
// ---------------------------------------------------------------------------
__global__ void pack_w(__bf16* __restrict__ dst, const float* __restrict__ src,
                       int Ktot, int sN, int sK, int total) {
  int id = blockIdx.x * 256 + threadIdx.x;
  if (id >= total) return;
  int h    = id & 15;
  int lane = (id >> 4) & 31;
  int blk  = id >> 9;
  int kcs  = Ktot >> 5;
  int t    = blk / kcs;
  int kc   = blk - t * kcs;
  int n    = t * 16 + (lane & 15);
  int k    = kc * 32 + ((lane >> 4) << 3) + (h & 7) + ((h >> 3) << 4);
  dst[id]  = (__bf16)expf(src[n * sN + k * sK]);
}

__global__ void vexp(float* __restrict__ dst, const float* __restrict__ src, int n) {
  int i = blockIdx.x * 256 + threadIdx.x;
  if (i < n) dst[i] = expf(src[i]);
}

// ---------------------------------------------------------------------------
// fused forward + backward
// ---------------------------------------------------------------------------
__global__ __launch_bounds__(NW * 32)
void brenier_fused(const float* __restrict__ U, const __bf16* __restrict__ W,
                   const float* __restrict__ B0, const float* __restrict__ B1,
                   const float* __restrict__ B2, const float* __restrict__ B3,
                   const float* __restrict__ B4, const float* __restrict__ EWU4,
                   const float* __restrict__ EWZ4, float* __restrict__ OUT) {
  __shared__ alignas(16) __bf16 uA[MT][72];          // u tile, bf16, padded
  __shared__ alignas(16) __bf16 zb[2][MT][520];      // z / g ping-pong, padded
  __shared__ float g4s[MT];                          // lrelu'(a4) per row
  __shared__ unsigned msk[NW][2][32][24];            // sign masks, C-frag order

  const int tid   = threadIdx.x;
  const int lane  = tid & 31;
  const int w     = tid >> 5;
  const int hi    = lane >> 4;
  const int lo    = lane & 15;
  const int rowbase = blockIdx.x * MT;   // first global batch row of block
  const int wrow0   = w * 32;            // first LDS row of this wave
  const int wrow1   = wrow0 + 16;        // second row-tile of this wave

  // ---- stage u tile (f32 -> bf16): 32 rows x 64 cols per wave ----------
  #pragma unroll 8
  for (int i = 0; i < 64; ++i) {
    int idx = i * 32 + lane;
    int r = idx >> 6, c = idx & 63;
    uA[wrow0 + r][c] = (__bf16)U[(size_t)(rowbase + wrow0 + r) * Dd + c];
  }
  __syncthreads();

  // ---- forward layer 0: z0 = lrelu(u E(wu0)^T + b0)^2 ------------------
  for (int t = 0; t < 32; ++t) {
    f32x8 acc0 = {}, acc1 = {};
    #pragma unroll
    for (int kc = 0; kc < 2; ++kc) {
      bf16x16 b = ldB(W + OFF_WU0F + (t * 2 + kc) * 512, lane);
      acc0 = wmma_bf16(ldA(&uA[wrow0][0], 72, lane, kc), b, acc0);
      acc1 = wmma_bf16(ldA(&uA[wrow1][0], 72, lane, kc), b, acc1);
    }
    int col = t * 16 + lo;
    float bias = B0[col];
    #pragma unroll
    for (int r = 0; r < 8; ++r) {
      float a0 = acc0[r] + bias;
      float s0 = a0 > 0.f ? a0 : ALPHA * a0;
      zb[0][wrow0 + r + 8 * hi][col] = (__bf16)(s0 * s0);
      float a1 = acc1[r] + bias;
      float s1 = a1 > 0.f ? a1 : ALPHA * a1;
      zb[0][wrow1 + r + 8 * hi][col] = (__bf16)(s1 * s1);
    }
  }
  __syncthreads();

  // ---- forward layers 1..3 ---------------------------------------------
  {
    const int owu[3] = {OFF_WU1F, OFF_WU2F, OFF_WU3F};
    const int owz[3] = {OFF_WZ1F, OFF_WZ2F, OFF_WZ3F};
    const float* const Bias[3] = {B1, B2, B3};
    #pragma unroll
    for (int L = 0; L < 3; ++L) {
      const int src = L & 1, dst = 1 - src;
      unsigned mcur0 = 0, mcur1 = 0;
      for (int t = 0; t < 32; ++t) {
        f32x8 acc0 = {}, acc1 = {};
        #pragma unroll
        for (int kc = 0; kc < 2; ++kc) {
          bf16x16 b = ldB(W + owu[L] + (t * 2 + kc) * 512, lane);
          acc0 = wmma_bf16(ldA(&uA[wrow0][0], 72, lane, kc), b, acc0);
          acc1 = wmma_bf16(ldA(&uA[wrow1][0], 72, lane, kc), b, acc1);
        }
        #pragma unroll
        for (int kc = 0; kc < 16; ++kc) {
          bf16x16 b = ldB(W + owz[L] + (t * 16 + kc) * 512, lane);
          acc0 = wmma_bf16(ldA(&zb[src][wrow0][0], 520, lane, kc), b, acc0);
          acc1 = wmma_bf16(ldA(&zb[src][wrow1][0], 520, lane, kc), b, acc1);
        }
        int col = t * 16 + lo;
        float bias = Bias[L][col];
        unsigned bits0 = 0, bits1 = 0;
        #pragma unroll
        for (int r = 0; r < 8; ++r) {
          float a0 = acc0[r] + bias;
          bits0 |= (a0 > 0.f) ? (1u << r) : 0u;
          zb[dst][wrow0 + r + 8 * hi][col] = (__bf16)(a0 > 0.f ? a0 : ALPHA * a0);
          float a1 = acc1[r] + bias;
          bits1 |= (a1 > 0.f) ? (1u << r) : 0u;
          zb[dst][wrow1 + r + 8 * hi][col] = (__bf16)(a1 > 0.f ? a1 : ALPHA * a1);
        }
        mcur0 |= bits0 << (8 * (t & 3));
        mcur1 |= bits1 << (8 * (t & 3));
        if ((t & 3) == 3) {
          msk[w][0][lane][L * 8 + (t >> 2)] = mcur0;
          msk[w][1][lane][L * 8 + (t >> 2)] = mcur1;
          mcur0 = mcur1 = 0;
        }
      }
      __syncthreads();
    }
  }

  // ---- layer 4 (scalar head): a4 = u ewu4 + z3 ewz4 + b4 ---------------
  {
    int row = lane;                       // 32 lanes, 32 rows: one each
    float p = 0.f;
    for (int d = 0; d < 64; ++d)
      p += (float)uA[wrow0 + row][d] * EWU4[d];
    for (int h = 0; h < 512; ++h)
      p += (float)zb[1][wrow0 + row][h] * EWZ4[h];
    float a4 = p + B4[0];
    g4s[wrow0 + row] = (a4 > 0.f) ? 1.f : ALPHA;
  }
  __syncthreads();

  // ---- backward ---------------------------------------------------------
  f32x8 gacc0[4] = {}, gacc1[4] = {};   // grad wrt u: 2 row-tiles x 4 n-tiles

  // g3 = (g4 * ewz4) .* lrelu'(a3)  -> zb[0]   (rank-1, elementwise)
  for (int t = 0; t < 32; ++t) {
    int col = t * 16 + lo;
    float wz4v = EWZ4[col];
    unsigned bits0 = (msk[w][0][lane][16 + (t >> 2)] >> (8 * (t & 3))) & 0xffu;
    unsigned bits1 = (msk[w][1][lane][16 + (t >> 2)] >> (8 * (t & 3))) & 0xffu;
    #pragma unroll
    for (int r = 0; r < 8; ++r) {
      int r0 = wrow0 + r + 8 * hi, r1 = wrow1 + r + 8 * hi;
      float dv0 = g4s[r0] * wz4v;
      zb[0][r0][col] = (__bf16)(((bits0 >> r) & 1) ? dv0 : ALPHA * dv0);
      float dv1 = g4s[r1] * wz4v;
      zb[0][r1][col] = (__bf16)(((bits1 >> r) & 1) ? dv1 : ALPHA * dv1);
    }
  }
  __syncthreads();

  {
    const int owut[4] = {0, OFF_WU1T, OFF_WU2T, OFF_WU3T};
    const int owzt[4] = {0, OFF_WZ1T, OFF_WZ2T, OFF_WZ3T};
    #pragma unroll
    for (int j = 3; j >= 1; --j) {
      const int gsrc = (j & 1) ^ 1;          // g3:0, g2:1, g1:0
      const int gdst = 1 - gsrc;
      // grad_u += g_j @ E(wu_j)   (K = 512, N = 64)
      #pragma unroll
      for (int nt = 0; nt < 4; ++nt) {
        #pragma unroll
        for (int kc = 0; kc < 16; ++kc) {
          bf16x16 b = ldB(W + owut[j] + (nt * 16 + kc) * 512, lane);
          gacc0[nt] = wmma_bf16(ldA(&zb[gsrc][wrow0][0], 520, lane, kc), b, gacc0[nt]);
          gacc1[nt] = wmma_bf16(ldA(&zb[gsrc][wrow1][0], 520, lane, kc), b, gacc1[nt]);
        }
      }
      // d_{j-1} = g_j @ E(wz_j); apply activation derivative -> g_{j-1}
      for (int t = 0; t < 32; ++t) {
        f32x8 acc0 = {}, acc1 = {};
        #pragma unroll
        for (int kc = 0; kc < 16; ++kc) {
          bf16x16 b = ldB(W + owzt[j] + (t * 16 + kc) * 512, lane);
          acc0 = wmma_bf16(ldA(&zb[gsrc][wrow0][0], 520, lane, kc), b, acc0);
          acc1 = wmma_bf16(ldA(&zb[gsrc][wrow1][0], 520, lane, kc), b, acc1);
        }
        int col = t * 16 + lo;
        if (j >= 2) {   // masks for a_{j-1} (layers 2,1)
          unsigned bits0 =
              (msk[w][0][lane][(j - 2) * 8 + (t >> 2)] >> (8 * (t & 3))) & 0xffu;
          unsigned bits1 =
              (msk[w][1][lane][(j - 2) * 8 + (t >> 2)] >> (8 * (t & 3))) & 0xffu;
          #pragma unroll
          for (int r = 0; r < 8; ++r) {
            float d0 = acc0[r];
            zb[gdst][wrow0 + r + 8 * hi][col] =
                (__bf16)(((bits0 >> r) & 1) ? d0 : ALPHA * d0);
            float d1 = acc1[r];
            zb[gdst][wrow1 + r + 8 * hi][col] =
                (__bf16)(((bits1 >> r) & 1) ? d1 : ALPHA * d1);
          }
        } else {        // j == 1: recompute layer-0 pre-acts; dz0/da0 = 2 s lrelu'
          f32x8 p0 = {}, p1 = {};
          #pragma unroll
          for (int kc = 0; kc < 2; ++kc) {
            bf16x16 b = ldB(W + OFF_WU0F + (t * 2 + kc) * 512, lane);
            p0 = wmma_bf16(ldA(&uA[wrow0][0], 72, lane, kc), b, p0);
            p1 = wmma_bf16(ldA(&uA[wrow1][0], 72, lane, kc), b, p1);
          }
          float bias = B0[col];
          #pragma unroll
          for (int r = 0; r < 8; ++r) {
            float a0 = p0[r] + bias;
            float s0 = a0 > 0.f ? a0 : ALPHA * a0;
            float dr0 = 2.f * s0 * (a0 > 0.f ? 1.f : ALPHA);
            zb[gdst][wrow0 + r + 8 * hi][col] = (__bf16)(acc0[r] * dr0);
            float a1 = p1[r] + bias;
            float s1 = a1 > 0.f ? a1 : ALPHA * a1;
            float dr1 = 2.f * s1 * (a1 > 0.f ? 1.f : ALPHA);
            zb[gdst][wrow1 + r + 8 * hi][col] = (__bf16)(acc1[r] * dr1);
          }
        }
      }
      __syncthreads();
    }
  }

  // grad_u += g0 @ E(wu0)    (g0 is in zb[1])
  #pragma unroll
  for (int nt = 0; nt < 4; ++nt) {
    #pragma unroll
    for (int kc = 0; kc < 16; ++kc) {
      bf16x16 b = ldB(W + OFF_WU0T + (nt * 16 + kc) * 512, lane);
      gacc0[nt] = wmma_bf16(ldA(&zb[1][wrow0][0], 520, lane, kc), b, gacc0[nt]);
      gacc1[nt] = wmma_bf16(ldA(&zb[1][wrow1][0], 520, lane, kc), b, gacc1[nt]);
    }
  }

  // add final-layer u-path (g4 * ewu4) and store grad
  #pragma unroll
  for (int nt = 0; nt < 4; ++nt) {
    int col = nt * 16 + lo;
    float w4 = EWU4[col];
    #pragma unroll
    for (int r = 0; r < 8; ++r) {
      int r0 = wrow0 + r + 8 * hi, r1 = wrow1 + r + 8 * hi;
      OUT[(size_t)(rowbase + r0) * Dd + col] = gacc0[nt][r] + g4s[r0] * w4;
      OUT[(size_t)(rowbase + r1) * Dd + col] = gacc1[nt][r] + g4s[r1] * w4;
    }
  }
}

// ---------------------------------------------------------------------------
// launch
// ---------------------------------------------------------------------------
extern "C" void kernel_launch(void* const* d_in, const int* in_sizes, int n_in,
                              void* d_out, int out_size, void* d_ws, size_t ws_size,
                              hipStream_t stream) {
  (void)in_sizes; (void)n_in; (void)out_size; (void)ws_size;
  const float* u   = (const float*)d_in[0];
  const float* wu0 = (const float*)d_in[1];
  const float* b0  = (const float*)d_in[2];
  const float* wu1 = (const float*)d_in[3];
  const float* wz1 = (const float*)d_in[4];
  const float* b1  = (const float*)d_in[5];
  const float* wu2 = (const float*)d_in[6];
  const float* wz2 = (const float*)d_in[7];
  const float* b2  = (const float*)d_in[8];
  const float* wu3 = (const float*)d_in[9];
  const float* wz3 = (const float*)d_in[10];
  const float* b3  = (const float*)d_in[11];
  const float* wu4 = (const float*)d_in[12];
  const float* wz4 = (const float*)d_in[13];
  const float* b4  = (const float*)d_in[14];

  __bf16* W   = (__bf16*)d_ws;
  float* ewu4 = (float*)((char*)d_ws + (size_t)W_END * 2);
  float* ewz4 = ewu4 + 64;

  // forward packs (B element (k,n) = w[n][k]):  sN = rowlen, sK = 1
  pack_w<<<(WU_SZ + 255) / 256, 256, 0, stream>>>(W + OFF_WU0F, wu0, 64, 64, 1, WU_SZ);
  pack_w<<<(WU_SZ + 255) / 256, 256, 0, stream>>>(W + OFF_WU1F, wu1, 64, 64, 1, WU_SZ);
  pack_w<<<(WU_SZ + 255) / 256, 256, 0, stream>>>(W + OFF_WU2F, wu2, 64, 64, 1, WU_SZ);
  pack_w<<<(WU_SZ + 255) / 256, 256, 0, stream>>>(W + OFF_WU3F, wu3, 64, 64, 1, WU_SZ);
  pack_w<<<(WZ_SZ + 255) / 256, 256, 0, stream>>>(W + OFF_WZ1F, wz1, 512, 512, 1, WZ_SZ);
  pack_w<<<(WZ_SZ + 255) / 256, 256, 0, stream>>>(W + OFF_WZ2F, wz2, 512, 512, 1, WZ_SZ);
  pack_w<<<(WZ_SZ + 255) / 256, 256, 0, stream>>>(W + OFF_WZ3F, wz3, 512, 512, 1, WZ_SZ);
  // backward (transposed) packs (B element (k,n) = w[k][n]): sN = 1, sK = rowlen
  pack_w<<<(WZ_SZ + 255) / 256, 256, 0, stream>>>(W + OFF_WZ1T, wz1, 512, 1, 512, WZ_SZ);
  pack_w<<<(WZ_SZ + 255) / 256, 256, 0, stream>>>(W + OFF_WZ2T, wz2, 512, 1, 512, WZ_SZ);
  pack_w<<<(WZ_SZ + 255) / 256, 256, 0, stream>>>(W + OFF_WZ3T, wz3, 512, 1, 512, WZ_SZ);
  pack_w<<<(WU_SZ + 255) / 256, 256, 0, stream>>>(W + OFF_WU0T, wu0, 512, 1, 64, WU_SZ);
  pack_w<<<(WU_SZ + 255) / 256, 256, 0, stream>>>(W + OFF_WU1T, wu1, 512, 1, 64, WU_SZ);
  pack_w<<<(WU_SZ + 255) / 256, 256, 0, stream>>>(W + OFF_WU2T, wu2, 512, 1, 64, WU_SZ);
  pack_w<<<(WU_SZ + 255) / 256, 256, 0, stream>>>(W + OFF_WU3T, wu3, 512, 1, 64, WU_SZ);
  vexp<<<1, 256, 0, stream>>>(ewu4, wu4, 64);
  vexp<<<2, 256, 0, stream>>>(ewz4, wz4, 512);

  brenier_fused<<<Bsz / MT, NW * 32, 0, stream>>>(u, W, b0, b1, b2, b3, b4,
                                                  ewu4, ewz4, (float*)d_out);
}
